// MethodCFGEncoder_64665027608673
// MI455X (gfx1250) — matrix operand order repaired and checked by reading, not compiled, not run on hardware
//
#include <hip/hip_runtime.h>
#include <hip/hip_bf16.h>

// ---------------------------------------------------------------------------
// MethodCFGEncoder forward for MI455X (gfx1250, wave32, WMMA).
// All matmuls lowered to v_wmma_f32_16x16x32_bf16 (fp32 accumulate).
// Dims: N_EXPR=20000, MAX_TOK=16, D=256, N_SYM=30000, N_OCC=100000,
//       N_NODES=50000, N_PATHS=1024, PLEN=32.  All GEMM M dims % 16 == 0.
// ---------------------------------------------------------------------------

#define N_EXPR   20000
#define MAX_TOK  16
#define DD       256
#define N_SYM    30000
#define N_OCC    100000
#define N_NODES  50000
#define N_PATHS  1024
#define PLEN     32
#define TD       768      // 3*D for GRU gates

typedef __attribute__((ext_vector_type(16))) __bf16 v16bf;
typedef __attribute__((ext_vector_type(8)))  __bf16 v8bf;
typedef __attribute__((ext_vector_type(8)))  float  v8f;

static __device__ inline v8f vzero() {
  v8f v = {0.f,0.f,0.f,0.f,0.f,0.f,0.f,0.f};
  return v;
}

static __device__ inline v8f wmma_bf16(v16bf a, v16bf b, v8f c) {
  // D = A(16x32 bf16) * B(32x16 bf16) + C(16x16 f32)
  return __builtin_amdgcn_wmma_f32_16x16x32_bf16(false, a, false, b,
                                                 (short)0, c, false, false);
}

// A fragment (16x32 bf16, row-major tile in LDS).  Per ISA layout:
// lanes 0-15 hold M=lane with Kloc = {0..7, 16..23}; lanes 16-31 hold
// M=lane-16 with Kloc = {8..15, 24..31}.  rowPtr = &tile[M][kt*32].
static __device__ inline v16bf load_a_frag(const __bf16* rowPtr, int lane) {
  int h8 = (lane >> 4) << 3;                       // 0 or 8
  v8bf lo = *(const v8bf*)(rowPtr + h8);           // Kloc h8 .. h8+7
  v8bf hi = *(const v8bf*)(rowPtr + 16 + h8);      // Kloc 16+h8 .. 23+h8
  v16bf a;
#pragma unroll
  for (int i = 0; i < 8; ++i) { a[i] = lo[i]; a[i + 8] = hi[i]; }
  return a;
}

// B fragment from pre-packed bf16 weights: for tile (kt,nt) each lane's 16
// halves are contiguous (32 bytes) => single vector load.
static __device__ inline v16bf load_b_frag(const __bf16* packed, int kt,
                                           int nt, int ntTiles, int lane) {
  return *(const v16bf*)(packed + (((size_t)kt * ntTiles + nt) * 32 + lane) * 16);
}

static __device__ inline float sigmoidf_(float x) {
  return 1.0f / (1.0f + __expf(-x));
}

// order-preserving float <-> uint encoding for atomicMax-based segment max
static __device__ inline unsigned encf(float f) {
  unsigned u = __float_as_uint(f);
  return (u & 0x80000000u) ? ~u : (u | 0x80000000u);
}
static __device__ inline float decf(unsigned e) {
  return (e & 0x80000000u) ? __uint_as_float(e & 0x7FFFFFFFu)
                           : __uint_as_float(~e);
}

// ---------------------------------------------------------------------------
// Weight pre-pack: f32 [K,N] row-major -> bf16 fragment-packed.
// B-matrix 32x16 bf16 layout: lane = N (mod 16); element i -> Kloc = 8*(lane>=16) + (i&7) + 16*(i>=8).
// ---------------------------------------------------------------------------
__global__ __launch_bounds__(256) void pack_b_kernel(const float* __restrict__ src,
                                                     __bf16* __restrict__ dst,
                                                     int K, int N) {
  int tile = blockIdx.x * 8 + (threadIdx.x >> 5);
  int lane = threadIdx.x & 31;
  int ntTiles = N >> 4;
  int totTiles = (K >> 5) * ntTiles;
  if (tile >= totTiles) return;
  int kt = tile / ntTiles, nt = tile % ntTiles;
  int col   = nt * 16 + (lane & 15);
  int kbase = kt * 32 + ((lane >> 4) << 3);        // +8 for lanes 16-31
  __bf16* d = dst + ((size_t)tile * 32 + lane) * 16;
#pragma unroll
  for (int i = 0; i < 16; ++i)
    d[i] = (__bf16)src[(size_t)(kbase + (i & 7) + ((i >> 3) << 4)) * N + col];
}

// ---------------------------------------------------------------------------
// Occurrence index resolution: flat_idx + deterministic last-writer claim.
// ---------------------------------------------------------------------------
__global__ void occ_index_kernel(const int* __restrict__ sym_expr,
                                 const int* __restrict__ sym_tok,
                                 int* __restrict__ flat_idx,
                                 int* __restrict__ claim) {
  int o = blockIdx.x * blockDim.x + threadIdx.x;
  if (o >= N_OCC) return;
  int fi = sym_expr[o] * MAX_TOK + sym_tok[o];
  flat_idx[o] = fi;
  atomicMax(&claim[fi], o);                        // last occurrence wins
}

// ---------------------------------------------------------------------------
// Generic gate:  cat=[prev|upd] (M x 512) ;  f=sigmoid(cat@Wf+bf) ;
// u=relu(cat@Wp+bp) ; out = f*prev + (1-f)*u.   One block = 16 rows x 256 cols.
// 8 waves x 2 n-tiles, K=512 -> 16 wmma k-steps, 64 wmma per wave.
// ---------------------------------------------------------------------------
__global__ __launch_bounds__(256) void gate_kernel(
    const float* __restrict__ prevBase, const int* __restrict__ prevIdx,
    const float* __restrict__ updBase,  const int* __restrict__ updIdx,
    const __bf16* __restrict__ WfP, const float* __restrict__ bfv,
    const __bf16* __restrict__ WpP, const float* __restrict__ bpv,
    float* __restrict__ outBase, const int* __restrict__ outIdx) {
  __shared__ __align__(16) __bf16 catB[16][520];   // 16 rows x 512 (+8 pad)
  __shared__ float prevS[16][DD];

  const int m0  = blockIdx.x * 16;
  const int tid = threadIdx.x;

  // stage cat tile (bf16) + prev tile (f32)
  {
    int row = tid >> 4, chunk = tid & 15;
    if (chunk < 8) {
      int pr = prevIdx ? prevIdx[m0 + row] : (m0 + row);
      const float* s = prevBase + (size_t)pr * DD + chunk * 32;
#pragma unroll
      for (int j = 0; j < 32; ++j) {
        float v = s[j];
        prevS[row][chunk * 32 + j] = v;
        catB[row][chunk * 32 + j]  = (__bf16)v;
      }
    } else {
      int ur = updIdx ? updIdx[m0 + row] : (m0 + row);
      const float* s = updBase + (size_t)ur * DD + (chunk - 8) * 32;
#pragma unroll
      for (int j = 0; j < 32; ++j)
        catB[row][256 + (chunk - 8) * 32 + j] = (__bf16)s[j];
    }
  }
  __syncthreads();

  const int wave = tid >> 5, lane = tid & 31;
  const int half = lane >> 4, nl = lane & 15;

  v8f accF[2], accU[2];
  accF[0] = vzero(); accF[1] = vzero();
  accU[0] = vzero(); accU[1] = vzero();

  for (int kt = 0; kt < 16; ++kt) {
    v16bf a = load_a_frag(&catB[nl][kt * 32], lane);
#pragma unroll
    for (int j = 0; j < 2; ++j) {
      int nt = wave * 2 + j;
      v16bf bF = load_b_frag(WfP, kt, nt, 16, lane);
      v16bf bU = load_b_frag(WpP, kt, nt, 16, lane);
      accF[j] = wmma_bf16(a, bF, accF[j]);
      accU[j] = wmma_bf16(a, bU, accU[j]);
    }
  }

#pragma unroll
  for (int j = 0; j < 2; ++j) {
    int col = (wave * 2 + j) * 16 + nl;
    float bfc = bfv[col], bpc = bpv[col];
#pragma unroll
    for (int rr = 0; rr < 8; ++rr) {
      int Mi = rr + 8 * half;
      float f = sigmoidf_(accF[j][rr] + bfc);
      float u = accU[j][rr] + bpc;  u = u > 0.f ? u : 0.f;
      float p = prevS[Mi][col];
      float o = f * p + (1.f - f) * u;
      int orow = outIdx ? outIdx[m0 + Mi] : (m0 + Mi);
      outBase[(size_t)orow * DD + col] = o;
    }
  }
}

// ---------------------------------------------------------------------------
// Mean pooling with claim indirection (resolves the token scatter lazily).
// ---------------------------------------------------------------------------
__global__ __launch_bounds__(256) void mean_kernel(const float* __restrict__ expr,
                                                   const float* __restrict__ occ_upd,
                                                   const int* __restrict__ claim,
                                                   float* __restrict__ combined) {
  int e = blockIdx.x, d = threadIdx.x;
  float s = 0.f;
#pragma unroll
  for (int t = 0; t < MAX_TOK; ++t) {
    int slot = e * MAX_TOK + t;
    int j = claim[slot];
    s += (j >= 0) ? occ_upd[(size_t)j * DD + d] : expr[(size_t)slot * DD + d];
  }
  combined[(size_t)e * DD + d] = s * (1.f / 16.f);
}

// ---------------------------------------------------------------------------
// GRU over woven [pn, pe] sequences, T=64 steps.  One block owns 16 paths.
// h kept in registers (lane-owned 16x16 sub-tiles); shared bf16 h-tile feeds
// the WMMA A operand each step.  96 wmma per wave per step.
// ---------------------------------------------------------------------------
__global__ __launch_bounds__(256) void gru_kernel(
    const float* __restrict__ pn, const float* __restrict__ pe,
    const int* __restrict__ plen,
    const __bf16* __restrict__ WxP, const __bf16* __restrict__ WhP,
    const float* __restrict__ bx, const float* __restrict__ bh,
    float* __restrict__ noe) {
  __shared__ __align__(16) __bf16 xB[16][264];
  __shared__ __align__(16) __bf16 hB[16][264];

  const int tid = threadIdx.x, wave = tid >> 5, lane = tid & 31;
  const int half = lane >> 4, nl = lane & 15;
  const int p0 = blockIdx.x * 16;

  for (int i = tid; i < 16 * 264; i += 256) (&hB[0][0])[i] = (__bf16)0.0f;

  float hreg[2][8];
#pragma unroll
  for (int j = 0; j < 2; ++j)
#pragma unroll
    for (int rr = 0; rr < 8; ++rr) hreg[j][rr] = 0.f;

  const int srow = tid >> 4, scol = (tid & 15) * 16;
  const int slen = plen[p0 + srow];

  for (int t = 0; t < 2 * PLEN; ++t) {
    __syncthreads();
    // stage x_t = (t even ? pn : pe)[path][t/2] masked by length
    {
      int l = t >> 1;
      bool valid = l < slen;
      const float* src = (t & 1) ? pe : pn;
      const float* sp = src + (((size_t)(p0 + srow)) * PLEN + l) * DD + scol;
#pragma unroll
      for (int j2 = 0; j2 < 16; ++j2)
        xB[srow][scol + j2] = (__bf16)(valid ? sp[j2] : 0.0f);
    }
    __syncthreads();

    // gx = x@Wx, gh = h@Wh  (thirds r,z,n handled per wave so elementwise is local)
    v8f ax[3][2], ah[3][2];
#pragma unroll
    for (int s = 0; s < 3; ++s)
#pragma unroll
      for (int j = 0; j < 2; ++j) { ax[s][j] = vzero(); ah[s][j] = vzero(); }

    for (int kt = 0; kt < 8; ++kt) {
      v16bf aX = load_a_frag(&xB[nl][kt * 32], lane);
      v16bf aH = load_a_frag(&hB[nl][kt * 32], lane);
#pragma unroll
      for (int j = 0; j < 2; ++j) {
        int ntb = wave * 2 + j;
#pragma unroll
        for (int s = 0; s < 3; ++s) {
          v16bf bX = load_b_frag(WxP, kt, s * 16 + ntb, 48, lane);
          v16bf bH = load_b_frag(WhP, kt, s * 16 + ntb, 48, lane);
          ax[s][j] = wmma_bf16(aX, bX, ax[s][j]);
          ah[s][j] = wmma_bf16(aH, bH, ah[s][j]);
        }
      }
    }

    // elementwise GRU cell on lane-owned elements
#pragma unroll
    for (int j = 0; j < 2; ++j) {
      int col = (wave * 2 + j) * 16 + nl;
      float bxr = bx[col], bxz = bx[256 + col], bxn = bx[512 + col];
      float bhr = bh[col], bhz = bh[256 + col], bhn = bh[512 + col];
#pragma unroll
      for (int rr = 0; rr < 8; ++rr) {
        float r = sigmoidf_((ax[0][j][rr] + bxr) + (ah[0][j][rr] + bhr));
        float z = sigmoidf_((ax[1][j][rr] + bxz) + (ah[1][j][rr] + bhz));
        float n = tanhf((ax[2][j][rr] + bxn) + r * (ah[2][j][rr] + bhn));
        float h = (1.f - z) * n + z * hreg[j][rr];
        hreg[j][rr] = h;
        if (!(t & 1)) {                      // node slot output (even steps)
          int Mi = rr + 8 * half;
          noe[(((size_t)(p0 + Mi)) * PLEN + (t >> 1)) * DD + col] = h;
        }
      }
    }
    __syncthreads();                         // all waves done reading hB
#pragma unroll
    for (int j = 0; j < 2; ++j) {
      int col = (wave * 2 + j) * 16 + nl;
#pragma unroll
      for (int rr = 0; rr < 8; ++rr)
        hB[rr + 8 * half][col] = (__bf16)hreg[j][rr];
    }
  }
}

// ---------------------------------------------------------------------------
// Attention scatter: per-occurrence scores + segment max (wave per row).
// ---------------------------------------------------------------------------
__global__ __launch_bounds__(256) void score_kernel(const float* __restrict__ noe,
                                                    const float* __restrict__ nodes,
                                                    const int* __restrict__ pnidx,
                                                    const int* __restrict__ plen,
                                                    float* __restrict__ scores,
                                                    unsigned* __restrict__ smax) {
  int o = blockIdx.x * 8 + (threadIdx.x >> 5);
  int lane = threadIdx.x & 31;
  int idx = pnidx[o];
  const float* x = noe + (size_t)o * DD;
  const float* nd = nodes + (size_t)idx * DD;
  float part = 0.f;
#pragma unroll
  for (int k = 0; k < 8; ++k) part += x[lane * 8 + k] * nd[lane * 8 + k];
  for (int off = 16; off > 0; off >>= 1) part += __shfl_down(part, off, 32);
  if (lane == 0) {
    float s = part * (1.0f / 16.0f);          // 1/sqrt(256)
    scores[o] = s;
    int path = o >> 5, l = o & 31;
    if (l < plen[path]) atomicMax(&smax[idx], encf(s));
  }
}

__global__ __launch_bounds__(256) void scatter_kernel(const float* __restrict__ noe,
                                                      const float* __restrict__ scores,
                                                      const unsigned* __restrict__ smax,
                                                      const int* __restrict__ pnidx,
                                                      const int* __restrict__ plen,
                                                      float* __restrict__ denom,
                                                      float* __restrict__ num) {
  int o = blockIdx.x * 8 + (threadIdx.x >> 5);
  int lane = threadIdx.x & 31;
  int path = o >> 5, l = o & 31;
  if (l >= plen[path]) return;                 // masked: e == 0, no contribution
  int idx = pnidx[o];
  float e = __expf(scores[o] - decf(smax[idx]));
  if (lane == 0) atomicAdd(&denom[idx], e);
  const float* x = noe + (size_t)o * DD;
  float* np = num + (size_t)idx * DD;
#pragma unroll
  for (int k = 0; k < 8; ++k)
    atomicAdd(&np[lane * 8 + k], e * x[lane * 8 + k]);
}

__global__ __launch_bounds__(256) void norm_kernel(float* __restrict__ num,
                                                   const float* __restrict__ denom) {
  int n = blockIdx.x, d = threadIdx.x;
  num[(size_t)n * DD + d] /= fmaxf(denom[n], 1e-9f);
}

// ---------------------------------------------------------------------------
// Host-side orchestration.  Input order (setup_inputs dict order, params last,
// nested dicts flattened in insertion order):
//  0 expressions 1 symbols 2 cfg_nodes 3 path_nodes 4 path_edges
//  5 sym_expr_idx 6 sym_tok_idx 7 sym_sym_idx 8 path_node_idx 9 path_lengths
//  10..13 sym_gate{Wf,bf,Wp,bp} 14..17 expr_gate 18..21 path_gate
//  22..25 scatter_gate 26 gru.Wx 27 gru.Wh 28 gru.bx 29 gru.bh
// ---------------------------------------------------------------------------
extern "C" void kernel_launch(void* const* d_in, const int* in_sizes, int n_in,
                              void* d_out, int out_size, void* d_ws, size_t ws_size,
                              hipStream_t stream) {
  (void)in_sizes; (void)n_in; (void)out_size; (void)ws_size;

  const float* expr  = (const float*)d_in[0];
  const float* syms  = (const float*)d_in[1];
  const float* cfg   = (const float*)d_in[2];
  const float* pnode = (const float*)d_in[3];
  const float* pedge = (const float*)d_in[4];
  const int* sym_expr = (const int*)d_in[5];
  const int* sym_tok  = (const int*)d_in[6];
  const int* sym_sym  = (const int*)d_in[7];
  const int* pnidx    = (const int*)d_in[8];
  const int* plen     = (const int*)d_in[9];

  const float* W[10] = {
      (const float*)d_in[10], (const float*)d_in[12],   // sym_gate Wf, Wp
      (const float*)d_in[14], (const float*)d_in[16],   // expr_gate
      (const float*)d_in[18], (const float*)d_in[20],   // path_gate
      (const float*)d_in[22], (const float*)d_in[24],   // scatter_gate
      (const float*)d_in[26], (const float*)d_in[27]};  // gru Wx, Wh
  const float* b_sg_f = (const float*)d_in[11]; const float* b_sg_p = (const float*)d_in[13];
  const float* b_eg_f = (const float*)d_in[15]; const float* b_eg_p = (const float*)d_in[17];
  const float* b_pg_f = (const float*)d_in[19]; const float* b_pg_p = (const float*)d_in[21];
  const float* b_cg_f = (const float*)d_in[23]; const float* b_cg_p = (const float*)d_in[25];
  const float* gbx = (const float*)d_in[28]; const float* gbh = (const float*)d_in[29];

  float* out = (float*)d_out;

  // workspace carve-out (256B aligned slots)
  char* wsp = (char*)d_ws;
  size_t off = 0;
  auto alloc = [&](size_t bytes) -> void* {
    void* p = wsp + off;
    off += (bytes + 255) & ~(size_t)255;
    return p;
  };
  float*  occ_upd  = (float*)alloc((size_t)N_OCC * DD * 4);
  float*  nodes_ws = (float*)alloc((size_t)N_NODES * DD * 4);
  float*  combined = (float*)alloc((size_t)N_EXPR * DD * 4);
  float*  pn_buf   = (float*)alloc((size_t)N_PATHS * PLEN * DD * 4);
  float*  noe      = (float*)alloc((size_t)N_PATHS * PLEN * DD * 4);
  float*  num      = (float*)alloc((size_t)N_NODES * DD * 4);
  float*  denom    = (float*)alloc((size_t)N_NODES * 4);
  unsigned* smax   = (unsigned*)alloc((size_t)N_NODES * 4);
  float*  scores   = (float*)alloc((size_t)N_PATHS * PLEN * 4);
  int*    claim    = (int*)alloc((size_t)N_EXPR * MAX_TOK * 4);
  int*    flat_idx = (int*)alloc((size_t)N_OCC * 4);
  __bf16* Wpk[10];
  for (int i = 0; i < 8; ++i) Wpk[i] = (__bf16*)alloc((size_t)512 * 256 * 2);
  for (int i = 8; i < 10; ++i) Wpk[i] = (__bf16*)alloc((size_t)256 * TD * 2);

  // init
  hipMemsetAsync(claim, 0xFF, (size_t)N_EXPR * MAX_TOK * 4, stream);  // -1
  hipMemsetAsync(num,   0,    (size_t)N_NODES * DD * 4, stream);
  hipMemsetAsync(denom, 0,    (size_t)N_NODES * 4, stream);
  hipMemsetAsync(smax,  0,    (size_t)N_NODES * 4, stream);           // min encoded

  // pack weights to bf16 fragment layout
  for (int i = 0; i < 8; ++i)
    pack_b_kernel<<<32, 256, 0, stream>>>(W[i], Wpk[i], 512, 256);    // 256 tiles
  for (int i = 8; i < 10; ++i)
    pack_b_kernel<<<48, 256, 0, stream>>>(W[i], Wpk[i], 256, TD);     // 384 tiles

  // occurrence indices + claim map
  occ_index_kernel<<<(N_OCC + 255) / 256, 256, 0, stream>>>(sym_expr, sym_tok,
                                                            flat_idx, claim);

  // 1) sym_gate over occurrences: prev = expr_flat[flat_idx], upd = symbols[sym_sym]
  gate_kernel<<<N_OCC / 16, 256, 0, stream>>>(
      expr, flat_idx, syms, sym_sym,
      Wpk[0], b_sg_f, Wpk[1], b_sg_p,
      occ_upd, nullptr);

  // 2) mean pool tokens (with scatter resolved via claim)
  mean_kernel<<<N_EXPR, 256, 0, stream>>>(expr, occ_upd, claim, combined);

  // nodes working copy
  hipMemcpyAsync(nodes_ws, cfg, (size_t)N_NODES * DD * 4,
                 hipMemcpyDeviceToDevice, stream);

  // 3) expr_gate: nodes[:N_EXPR] = gate(nodes[:N_EXPR], combined)
  gate_kernel<<<N_EXPR / 16, 256, 0, stream>>>(
      cfg, nullptr, combined, nullptr,
      Wpk[2], b_eg_f, Wpk[3], b_eg_p,
      nodes_ws, nullptr);

  // 4) path_gate: pn = gate(path_nodes, nodes[path_node_idx])
  gate_kernel<<<(N_PATHS * PLEN) / 16, 256, 0, stream>>>(
      pnode, nullptr, nodes_ws, pnidx,
      Wpk[4], b_pg_f, Wpk[5], b_pg_p,
      pn_buf, nullptr);

  // 5) GRU over woven (pn, pe) sequences; node-slot outputs -> noe
  gru_kernel<<<N_PATHS / 16, 256, 0, stream>>>(
      pn_buf, pedge, plen, Wpk[8], Wpk[9], gbx, gbh, noe);

  // 6) masked attention scatter to nodes
  score_kernel<<<(N_PATHS * PLEN) / 8, 256, 0, stream>>>(
      noe, nodes_ws, pnidx, plen, scores, smax);
  scatter_kernel<<<(N_PATHS * PLEN) / 8, 256, 0, stream>>>(
      noe, scores, smax, pnidx, plen, denom, num);
  norm_kernel<<<N_NODES, 256, 0, stream>>>(num, denom);

  // 7) scatter_gate -> final output
  gate_kernel<<<N_NODES / 16, 256, 0, stream>>>(
      nodes_ws, nullptr, num, nullptr,
      Wpk[6], b_cg_f, Wpk[7], b_cg_p,
      out, nullptr);
}
